// FFTBias_8727373545479
// MI455X (gfx1250) — matrix operand order repaired and checked by reading, not compiled, not run on hardware
//
#include <hip/hip_runtime.h>
#include <hip/hip_bf16.h>

typedef __attribute__((ext_vector_type(16))) _Float16 v16h;
typedef __attribute__((ext_vector_type(8)))  float    v8f;

#define SEQ 2048
#define NH  16
#define NE  64
#define NB  4

// ---------------------------------------------------------------------------
// pbv[b,t',h,e] = sum_t w[h,|t'-t|] * v[b,t,h,e]
// One block = one (b,h), 128 output rows, all 64 columns. 8 waves,
// each wave owns one 16-row tile x four 16-col tiles (16x64 of output).
//
// Toeplitz symmetry: w[|c-K|] = wsym[2047 + K - c], wsym[j] = w[|j-2047|],
// so the A-fragment gather is two contiguous 8-half windows per lane.
// ---------------------------------------------------------------------------
__global__ __launch_bounds__(256)
void toeplitz_mm_kernel(const float* __restrict__ v,
                        const float* __restrict__ w,
                        float* __restrict__ pbv)
{
    __shared__ __align__(16) _Float16 wsym[4096];      // 8 KB: symmetric extension of w[h,:]
    __shared__ __align__(16) _Float16 vt[NE][40];      // 5 KB: transposed v chunk (80B rows)

    const int bh = blockIdx.y;          // 0..63
    const int b  = bh / NH;
    const int h  = bh % NH;
    const int I0base = blockIdx.x * 128;

    const int tid  = threadIdx.x;
    const int wave = tid >> 5;          // 0..7
    const int lane = tid & 31;
    const int m    = lane & 15;         // M for A, N for B/C/D
    const bool hi  = lane >= 16;

    // stage wsym[j] = w[h, |j - 2047|]  (f32 -> f16), one extra zero pad half
    for (int j = tid; j < 4095; j += 256) {
        int d = j - (SEQ - 1); d = d < 0 ? -d : d;
        wsym[j] = (_Float16)w[h * SEQ + d];
    }
    if (tid == 0) wsym[4095] = (_Float16)0.f;

    const int I0 = I0base + wave * 16;  // this wave's row tile
    const float* vbase = v + (((size_t)b * SEQ) * NH + h) * NE;  // + t*NH*NE + e
    const uint32_t* w32 = (const uint32_t*)wsym;

    v8f acc[4] = {};

    for (int t0 = 0; t0 < SEQ; t0 += 32) {
        __syncthreads();   // vt consumed by previous iteration
        // ---- stage V chunk transposed: vt[e][k] = v[b, t0+k, h, e], f32->f16
        {
            const int k  = tid >> 3;          // 0..31
            const int e0 = (tid & 7) * 8;     // 8 consecutive e
            const float* src = vbase + (size_t)(t0 + k) * (NH * NE) + e0;
            #pragma unroll
            for (int q = 0; q < 8; ++q)
                vt[e0 + q][k] = (_Float16)src[q];
            if (t0 + 32 < SEQ)
                __builtin_prefetch(src + 32 * NH * NE, 0, 3);  // next chunk, WGP scope
        }
        __syncthreads();

        // ---- A fragment (16x32 f16): A[i,kk] = wsym[2047 + (kk) - (I0+i-t0)]
        // lane layout: M=m; lanes 0-15: K(p)= p<8 ? p : p+8 ; lanes 16-31: +8
        v16h a;
        {
            const int kb     = hi ? 8 : 0;
            const int start0 = (SEQ - 1) + kb - (I0 + m - t0);   // >= 0, <= 4071
            const int s0     = start0 >> 1;                      // aligned dword index
            const uint32_t sh = (uint32_t)(start0 & 1) * 16u;

            uint32_t d0[5], d1[5];
            #pragma unroll
            for (int i = 0; i < 5; ++i) d0[i] = w32[s0 + i];         // halves [2s0, 2s0+9]
            #pragma unroll
            for (int i = 0; i < 5; ++i) d1[i] = w32[s0 + 8 + i];     // +16 halves, same parity

            union { v16h h; uint32_t u[8]; } au;
            #pragma unroll
            for (int i = 0; i < 4; ++i)
                au.u[i] = (uint32_t)(((((uint64_t)d0[i + 1]) << 32) | d0[i]) >> sh);
            #pragma unroll
            for (int i = 0; i < 4; ++i)
                au.u[4 + i] = (uint32_t)(((((uint64_t)d1[i + 1]) << 32) | d1[i]) >> sh);
            a = au.h;
        }

        // ---- 4 column tiles: B[k,j] = vt[J0+j][k]
        // B layout: N=m; lanes 0-15: K(p)= p<8 ? p : p+8 ; lanes 16-31: +8
        const int kb2 = hi ? 8 : 0;
        #pragma unroll
        for (int j = 0; j < 4; ++j) {
            v16h bf;
            #pragma unroll
            for (int p = 0; p < 16; ++p)
                bf[p] = vt[j * 16 + m][kb2 + p + (p >= 8 ? 8 : 0)];   // 2x contiguous 16B -> ds_load_b128
            acc[j] = __builtin_amdgcn_wmma_f32_16x16x32_f16(
                false, a, false, bf, (short)0, acc[j], false, false);
        }
    }

    // ---- store: D f32 layout: N = m; VGPR r -> M = r + (hi ? 8 : 0)
    #pragma unroll
    for (int j = 0; j < 4; ++j) {
        const int e = j * 16 + m;
        #pragma unroll
        for (int r = 0; r < 8; ++r) {
            const int tp = I0 + r + (hi ? 8 : 0);
            pbv[(((size_t)b * SEQ + tp) * NH + h) * NE + e] = acc[j][r];
        }
    }
}

// ---------------------------------------------------------------------------
// z_pb[t',h] = sum_t w[h,|t'-t|] * o[t]   (tiny: 134 MFLOP, plain f32)
// ---------------------------------------------------------------------------
__global__ __launch_bounds__(256)
void toeplitz_vec_kernel(const float* __restrict__ w,
                         const float* __restrict__ o,
                         float* __restrict__ zpb)
{
    __shared__ float wsh[SEQ];
    __shared__ float osh[SEQ];
    const int h = blockIdx.y;
    for (int i = threadIdx.x; i < SEQ; i += 256) {
        wsh[i] = w[h * SEQ + i];
        osh[i] = o[i];
    }
    __syncthreads();
    const int tp = blockIdx.x * 256 + threadIdx.x;
    float s = 0.f;
    #pragma unroll 8
    for (int t = 0; t < SEQ; ++t) {
        int d = tp - t; d = d < 0 ? -d : d;
        s = fmaf(wsh[d], osh[t], s);
    }
    zpb[tp * NH + h] = s;
}

extern "C" void kernel_launch(void* const* d_in, const int* in_sizes, int n_in,
                              void* d_out, int out_size, void* d_ws, size_t ws_size,
                              hipStream_t stream)
{
    const float* v = (const float*)d_in[0];   // (B,S,H,E)
    const float* w = (const float*)d_in[1];   // (1,H,S)
    const float* o = (const float*)d_in[2];   // (S,)
    float* pbv = (float*)d_out;                                   // (B,S,H,E) flat
    float* zpb = pbv + (size_t)NB * SEQ * NH * NE;                // (1,S,H) flat

    dim3 grid1(SEQ / 128, NB * NH);
    toeplitz_mm_kernel<<<grid1, 256, 0, stream>>>(v, w, pbv);

    dim3 grid2(SEQ / 256, NH);
    toeplitz_vec_kernel<<<grid2, 256, 0, stream>>>(w, o, zpb);
}
